// CharCondDiscriminatorAP_9466107920486
// MI455X (gfx1250) — matrix operand order repaired and checked by reading, not compiled, not run on hardware
//
#include <hip/hip_runtime.h>

typedef _Float16 h16;
typedef __attribute__((ext_vector_type(16))) _Float16 v16h;
typedef __attribute__((ext_vector_type(8)))  _Float16 v8h;
typedef __attribute__((ext_vector_type(8)))  float    v8f;

#define LEAK 0.1f

// ---------------------------------------------------------------------------
// zero fill (16B granules)
__global__ void zero_fill(uint4* p, long long n16) {
  for (long long i = (long long)blockIdx.x * blockDim.x + threadIdx.x; i < n16;
       i += (long long)gridDim.x * blockDim.x) {
    uint4 z; z.x = 0u; z.y = 0u; z.z = 0u; z.w = 0u;
    p[i] = z;
  }
}

// ---------------------------------------------------------------------------
// style1 = g_style @ sp1_w.T (2x32), style2 = g_style @ sp2_w.T (2x64) -> f16
__global__ void style_kernel(const float* __restrict__ g, const float* __restrict__ sp1,
                             const float* __restrict__ sp2, h16* __restrict__ sb) {
  int b = blockIdx.x;          // 0..1
  int j = threadIdx.x;         // 0..95
  const float* gr = g + b * 256;
  if (j < 32) {
    float a = 0.f;
    for (int s = 0; s < 256; ++s) a += gr[s] * sp1[j * 256 + s];
    sb[b * 32 + j] = (h16)a;
  } else {
    int j2 = j - 32;
    float a = 0.f;
    for (int s = 0; s < 256; ++s) a += gr[s] * sp2[j2 * 256 + s];
    sb[64 + b * 64 + j2] = (h16)a;
  }
}

// cls[n] = argmax_c label[t][b][c],  n = b*128 + t
__global__ void cls_kernel(const float* __restrict__ label, int* __restrict__ cls) {
  int n = threadIdx.x;         // 256 threads, 1 block
  int b = n >> 7, t = n & 127;
  const float* L = label + (size_t)(t * 2 + b) * 80;
  int best = 0; float bv = L[0];
  for (int c = 1; c < 80; ++c) { float v = L[c]; if (v > bv) { bv = v; best = c; } }
  cls[n] = best;
}

// OIHW f32 -> [Cout][(kh*3+kw)*Cinp + cin] f16, zero-padded cin>=Cin
__global__ void wxform_kernel(const float* __restrict__ w, h16* __restrict__ wT,
                              int Cin, int Cinp, int total) {
  for (int idx = blockIdx.x * blockDim.x + threadIdx.x; idx < total;
       idx += gridDim.x * blockDim.x) {
    int cin = idx % Cinp;
    int r   = idx / Cinp;
    int tap = r % 9;
    int co  = r / 9;
    float v = 0.f;
    if (cin < Cin) {
      int kh = tap / 3, kw = tap % 3;
      v = w[(((size_t)co * Cin + cin) * 3 + kh) * 3 + kw];
    }
    wT[idx] = (h16)v;
  }
}

// ---------------------------------------------------------------------------
// stem: 7x7 conv, Cin=1, pad (0,3), lrelu -> A0[b][oh][ow+1][co] (NHWC, Cp=96)
__global__ void conv0_kernel(const float* __restrict__ x, const float* __restrict__ w,
                             const float* __restrict__ bias, h16* __restrict__ A0, int total) {
  for (int idx = blockIdx.x * blockDim.x + threadIdx.x; idx < total;
       idx += gridDim.x * blockDim.x) {
    int ow = idx & 511; int r = idx >> 9;
    int co = r & 63;    r >>= 6;
    int oh = r % 58;    int b = r / 58;
    float acc = bias[co];
    for (int kh = 0; kh < 7; ++kh) {
      const float* xr = x + ((size_t)b * 64 + oh + kh) * 512;
      const float* wr = w + (co * 7 + kh) * 7;
      for (int kw = 0; kw < 7; ++kw) {
        int iw = ow + kw - 3;
        if (iw >= 0 && iw < 512) acc += xr[iw] * wr[kw];
      }
    }
    acc = acc > 0.f ? acc : acc * LEAK;
    A0[((((size_t)b * 58 + oh) * 514) + ow + 1) * 96 + co] = (h16)acc;
  }
}

// broadcast style1 (32 ch) into A0 channels 64..95
__global__ void fill_style1(const h16* __restrict__ sb, h16* __restrict__ A0, int total) {
  for (int idx = blockIdx.x * blockDim.x + threadIdx.x; idx < total;
       idx += gridDim.x * blockDim.x) {
    int j = idx & 31; int r = idx >> 5;
    int w = r & 511;  r >>= 9;
    int oh = r % 58;  int b = r / 58;
    A0[(((size_t)b * 58 + oh) * 514 + w + 1) * 96 + 64 + j] = sb[b * 32 + j];
  }
}

// broadcast style2 (64 ch) into A2 channels 128..191
__global__ void fill_style2(const h16* __restrict__ sb, h16* __restrict__ A2, int total) {
  for (int idx = blockIdx.x * blockDim.x + threadIdx.x; idx < total;
       idx += gridDim.x * blockDim.x) {
    int j = idx & 63; int r = idx >> 6;
    int w = r & 255;  r >>= 8;
    int oh = r % 26;  int b = r / 26;
    A2[(((size_t)b * 26 + oh) * 258 + w + 1) * 192 + 128 + j] = sb[64 + b * 64 + j];
  }
}

// sal (spaced_style 32ch + label 80ch) into A3 channels 128..239
__global__ void fill_sal(const float* __restrict__ sstyle, const float* __restrict__ label,
                         h16* __restrict__ A3, int total) {
  for (int idx = blockIdx.x * blockDim.x + threadIdx.x; idx < total;
       idx += gridDim.x * blockDim.x) {
    int c = idx % 112; int r = idx / 112;
    int w = r & 127;   r >>= 7;
    int h = r % 12;    int b = r / 12;
    float v;
    if (c < 32) v = sstyle[(size_t)(w * 2 + b) * 32 + c];
    else        v = label[(size_t)(w * 2 + b) * 80 + (c - 32)];
    A3[(((size_t)b * 12 + h) * 130 + w + 1) * 256 + 128 + c] = (h16)v;
  }
}

// feat[n][128+j] = char_style[b][cls[n]][j]
__global__ void fill_cs(const float* __restrict__ cstyle, const int* __restrict__ cls,
                        float* __restrict__ feat, int total) {
  for (int idx = blockIdx.x * blockDim.x + threadIdx.x; idx < total;
       idx += gridDim.x * blockDim.x) {
    int j = idx & 31; int n = idx >> 5;
    int b = n >> 7;
    feat[(size_t)n * 160 + 128 + j] = cstyle[((size_t)b * 80 + cls[n]) * 32 + j];
  }
}

// ---------------------------------------------------------------------------
// 2x2 avg pool NHWC (f16), write into padded dest buffer at woff / Cp
__global__ void pool2_kernel(const h16* __restrict__ in, h16* __restrict__ out,
                             int Ho, int Wo, int C, int Wi, int Wop, int Cp, int woff,
                             int total) {
  for (int idx = blockIdx.x * blockDim.x + threadIdx.x; idx < total;
       idx += gridDim.x * blockDim.x) {
    int c = idx % C;  int r = idx / C;
    int ow = r % Wo;  r /= Wo;
    int oh = r % Ho;  int b = r / Ho;
    const h16* p = in + (((size_t)b * (2 * Ho) + 2 * oh) * Wi + 2 * ow) * C + c;
    float s = (float)p[0] + (float)p[C] + (float)p[(size_t)Wi * C] + (float)p[(size_t)Wi * C + C];
    out[(((size_t)b * Ho + oh) * Wop + ow + woff) * Cp + c] = (h16)(0.25f * s);
  }
}

// ---------------------------------------------------------------------------
// generic 3x3 conv (pad h=0, w=1 pre-materialized in padded input), lrelu.
// implicit GEMM on v_wmma_f32_16x16x32_f16.
// in : [B][Hin][Wp][Cinp]  f16  (w-padded, channel-padded to mult of 32)
// wT : [Cout][(kh*3+kw)*Cinp + cin] f16
// out: [B][OH][outWp][outCp] f16, tile written at w+outWoff, channels 0..Cout-1
// grid: (OW/16, OH, B), block: 32 * (Cout/16)
__global__ void conv3x3_wmma(const h16* __restrict__ in, const h16* __restrict__ wT,
                             const float* __restrict__ bias, h16* __restrict__ out,
                             int Hin, int Wp, int Cinp, int OH, int OW, int Cout,
                             int outWp, int outCp, int outWoff) {
  const int lane = threadIdx.x & 31;
  const int wave = threadIdx.x >> 5;
  const int row  = lane & 15;
  const int hi   = lane >> 4;
  const int owt  = blockIdx.x;
  const int oh   = blockIdx.y;
  const int b    = blockIdx.z;
  const int co_base = wave * 16;
  const int ow   = owt * 16 + row;      // A-matrix row M for this lane
  const int Kp   = 9 * Cinp;

  v8f acc = {};
  const h16* inB = in + (size_t)b * Hin * Wp * Cinp;
  const h16* wB  = wT + (size_t)(co_base + row) * Kp + hi * 8;

  for (int kh = 0; kh < 3; ++kh) {
    const h16* inRow = inB + (size_t)(oh + kh) * Wp * Cinp;
    for (int kw = 0; kw < 3; ++kw) {
      const h16* apos = inRow + (size_t)(ow + kw) * Cinp + hi * 8;
      const h16* bpos = wB + (kh * 3 + kw) * Cinp;
      for (int cb = 0; cb < Cinp; cb += 32) {
        union { v16h v; v8h h[2]; } A, Bm;
        A.h[0]  = *(const v8h*)(apos + cb);        // K = hi*8 + {0..7}
        A.h[1]  = *(const v8h*)(apos + cb + 16);   // K = hi*8 + {16..23}
        Bm.h[0] = *(const v8h*)(bpos + cb);        // K = hi*8 + {0..7}
        Bm.h[1] = *(const v8h*)(bpos + cb + 16);   // K = hi*8 + {16..23}
        acc = __builtin_amdgcn_wmma_f32_16x16x32_f16(false, A.v, false, Bm.v,
                                                     (short)0, acc, false, false);
      }
    }
  }

  const float bv = bias[co_base + row];
  size_t obase = (((size_t)b * OH + oh) * outWp + outWoff + (size_t)owt * 16) * outCp
               + co_base + row;
#pragma unroll
  for (int v = 0; v < 8; ++v) {
    int m = v + hi * 8;                            // output column (ow) offset
    float val = acc[v] + bv;
    val = val > 0.f ? val : val * LEAK;
    out[obase + (size_t)m * outCp] = (h16)val;
  }
}

// ---------------------------------------------------------------------------
// final feature-map head: pM[b][ow] = 3x3x256 dot on A5 + bias
__global__ void fm_kernel(const h16* __restrict__ A5, const float* __restrict__ fw,
                          const float* __restrict__ fb, float* __restrict__ out) {
  __shared__ float red[256];
  int blk = blockIdx.x;           // 0..127
  int b = blk >> 6, ow = blk & 63;
  float a = 0.f;
  for (int e = threadIdx.x; e < 2304; e += 256) {
    int kw = e % 3; int r = e / 3; int kh = r % 3; int c = r / 3;
    a += (float)A5[(((size_t)b * 3 + kh) * 68 + (ow + kw + 1)) * 256 + c] * fw[e];
  }
  red[threadIdx.x] = a;
  __syncthreads();
  for (int s = 128; s > 0; s >>= 1) {
    if (threadIdx.x < s) red[threadIdx.x] += red[threadIdx.x + s];
    __syncthreads();
  }
  if (threadIdx.x == 0) out[blk] = red[0] + fb[0];
}

// ---------------------------------------------------------------------------
// per-sample class-conditioned window conv as WMMA GEMM:
// M = 3 windows (rows 3..15 zero), N = 128 out-ch (8 waves x 16), K = 2304.
// A gathered from A5 (NHWC f16), B = cd_conv_w[cls] f32 -> f16 on the fly.
// grid: 256 blocks (n), block: 256 threads (8 waves)
__global__ void cd_conv_wmma(const h16* __restrict__ A5, const float* __restrict__ W,
                             const float* __restrict__ Wb, const int* __restrict__ cls,
                             float* __restrict__ feat) {
  const int n = blockIdx.x;
  const int b = n >> 7;
  const int t = n & 127;
  const int t2 = t >> 1;
  const int lane = threadIdx.x & 31;
  const int wave = threadIdx.x >> 5;
  const int row  = lane & 15;
  const int hi   = lane >> 4;
  const int co   = wave * 16 + row;
  const int c_   = cls[n];
  const float* wrow = W + ((size_t)c_ * 128 + co) * 2304;
  const int mm = (row < 3) ? row : 0;               // clamp window index
  const h16 msk = (h16)((row < 3) ? 1.0f : 0.0f);   // zero rows 3..15

  v8f acc = {};
  for (int kb = 0; kb < 2304; kb += 32) {
    union { v16h v; h16 e[16]; } A, Bm;
#pragma unroll
    for (int i = 0; i < 16; ++i) {
      int k = kb + hi * 8 + (i & 7) + ((i >> 3) << 4);   // A striping
      int c = k / 9; int rem = k - c * 9; int rr = rem / 3; int cw = rem - rr * 3;
      h16 v = A5[(((size_t)b * 3 + rr) * 68 + (t2 + mm + cw)) * 256 + c];
      A.e[i] = v * msk;
    }
#pragma unroll
    for (int i = 0; i < 16; ++i) Bm.e[i] = (h16)wrow[kb + hi * 8 + (i & 7) + ((i >> 3) << 4)];
    acc = __builtin_amdgcn_wmma_f32_16x16x32_f16(false, A.v, false, Bm.v,
                                                 (short)0, acc, false, false);
  }
  if (hi == 0) {                                    // rows 0..2 live in acc[0..2]
    float bv = Wb[c_ * 128 + co];
    float s = 0.f;
#pragma unroll
    for (int v = 0; v < 3; ++v) {
      float x = acc[v] + bv;
      s += (x > 0.f ? x : LEAK * x);
    }
    feat[(size_t)n * 160 + co] = s * (1.0f / 3.0f);
  }
}

// ---------------------------------------------------------------------------
// per-sample class FCs: h1 = relu(feat @ fc1w[cls] + b), pChar = h1 @ fc2w[cls] + b
__global__ void fc_kernel(const float* __restrict__ feat, const int* __restrict__ cls,
                          const float* __restrict__ w1, const float* __restrict__ b1,
                          const float* __restrict__ w2, const float* __restrict__ b2,
                          float* __restrict__ out) {
  __shared__ float fl[160];
  __shared__ float sh[128];
  int n = blockIdx.x;            // 0..255
  int o = threadIdx.x;           // 0..127
  int c_ = cls[n];
  for (int d = o; d < 160; d += 128) fl[d] = feat[(size_t)n * 160 + d];
  __syncthreads();
  float a = b1[c_ * 128 + o];
  for (int d = 0; d < 160; ++d) a += fl[d] * w1[((size_t)c_ * 160 + d) * 128 + o];
  a = a > 0.f ? a : 0.f;
  sh[o] = a * w2[c_ * 128 + o];
  __syncthreads();
  for (int s = 64; s > 0; s >>= 1) {
    if (o < s) sh[o] += sh[o + s];
    __syncthreads();
  }
  if (o == 0) out[128 + n] = sh[0] + b2[c_];
}

// ===========================================================================
extern "C" void kernel_launch(void* const* d_in, const int* in_sizes, int n_in,
                              void* d_out, int out_size, void* d_ws, size_t ws_size,
                              hipStream_t stream) {
  (void)in_sizes; (void)n_in; (void)out_size; (void)ws_size;
  const float* x      = (const float*)d_in[0];
  const float* label  = (const float*)d_in[1];
  const float* gstyle = (const float*)d_in[2];
  const float* sstyle = (const float*)d_in[3];
  const float* cstyle = (const float*)d_in[4];
  const float* sp1w   = (const float*)d_in[5];
  const float* sp2w   = (const float*)d_in[6];
  const float* in_w   = (const float*)d_in[7];
  const float* in_b   = (const float*)d_in[8];
  const float* c1w1   = (const float*)d_in[9];
  const float* c1b1   = (const float*)d_in[10];
  const float* c1w2   = (const float*)d_in[11];
  const float* c1b2   = (const float*)d_in[12];
  const float* c2w    = (const float*)d_in[13];
  const float* c2b    = (const float*)d_in[14];
  const float* c3w1   = (const float*)d_in[15];
  const float* c3b1   = (const float*)d_in[16];
  const float* c3w2   = (const float*)d_in[17];
  const float* c3b2   = (const float*)d_in[18];
  const float* fm_w   = (const float*)d_in[19];
  const float* fm_b   = (const float*)d_in[20];
  const float* cdw    = (const float*)d_in[21];
  const float* cdb    = (const float*)d_in[22];
  const float* f1w    = (const float*)d_in[23];
  const float* f1b    = (const float*)d_in[24];
  const float* f2w    = (const float*)d_in[25];
  const float* f2b    = (const float*)d_in[26];
  float* out = (float*)d_out;
  char* ws = (char*)d_ws;

  size_t off = 0;
  auto alloc = [&](size_t bytes) { size_t o = off; off = (off + bytes + 255) & ~(size_t)255; return o; };
  // zero-filled padded activation buffers (NHWC f16)
  size_t oA0 = alloc(2ull * 58 * 514 * 96 * 2);   // stem out + style1, wpad 1
  size_t oA1 = alloc(2ull * 28 * 258 * 64 * 2);   // pool1 out, wpad 1
  size_t oA2 = alloc(2ull * 26 * 258 * 192 * 2);  // c1w2 out + style2, wpad 1
  size_t oA3 = alloc(2ull * 12 * 130 * 256 * 2);  // pool2 out + sal, cpad 240->256, wpad 1
  size_t oA4 = alloc(2ull * 5 * 66 * 128 * 2);    // pool3 out, wpad 1
  size_t oA5 = alloc(2ull * 3 * 68 * 256 * 2);    // c3w2 out, wpad 2 (== mLp)
  size_t zeroBytes = off;
  // non-padded conv outputs (fully overwritten)
  size_t oB1 = alloc(2ull * 56 * 512 * 64 * 2);
  size_t oB2 = alloc(2ull * 24 * 256 * 128 * 2);
  size_t oB3 = alloc(2ull * 10 * 128 * 128 * 2);
  // transformed f16 weights
  size_t oW1 = alloc(64ull  * 9 * 96  * 2);
  size_t oW2 = alloc(128ull * 9 * 64  * 2);
  size_t oW3 = alloc(128ull * 9 * 192 * 2);
  size_t oW4 = alloc(128ull * 9 * 256 * 2);
  size_t oW5 = alloc(256ull * 9 * 128 * 2);
  size_t oSB = alloc(192 * 2);          // style1/style2 f16
  size_t oCL = alloc(256 * 4);          // cls ints
  size_t oFT = alloc(256ull * 160 * 4); // feat f32

  h16* A0 = (h16*)(ws + oA0); h16* A1 = (h16*)(ws + oA1); h16* A2 = (h16*)(ws + oA2);
  h16* A3 = (h16*)(ws + oA3); h16* A4 = (h16*)(ws + oA4); h16* A5 = (h16*)(ws + oA5);
  h16* B1 = (h16*)(ws + oB1); h16* B2 = (h16*)(ws + oB2); h16* B3 = (h16*)(ws + oB3);
  h16* W1 = (h16*)(ws + oW1); h16* W2 = (h16*)(ws + oW2); h16* W3 = (h16*)(ws + oW3);
  h16* W4 = (h16*)(ws + oW4); h16* W5 = (h16*)(ws + oW5);
  h16* SB = (h16*)(ws + oSB);
  int* CL = (int*)(ws + oCL);
  float* FT = (float*)(ws + oFT);

  auto blocks = [](long long total) { return (unsigned)((total + 255) / 256); };

  // 0) zero the padded activation region (determinism: pads/extra channels = 0)
  zero_fill<<<dim3(2048), dim3(256), 0, stream>>>((uint4*)ws, (long long)(zeroBytes / 16));

  // 1) small precomputations
  style_kernel<<<dim3(2), dim3(96), 0, stream>>>(gstyle, sp1w, sp2w, SB);
  cls_kernel<<<dim3(1), dim3(256), 0, stream>>>(label, CL);
  wxform_kernel<<<blocks(64LL * 9 * 96), 256, 0, stream>>>(c1w1, W1, 96, 96, 64 * 9 * 96);
  wxform_kernel<<<blocks(128LL * 9 * 64), 256, 0, stream>>>(c1w2, W2, 64, 64, 128 * 9 * 64);
  wxform_kernel<<<blocks(128LL * 9 * 192), 256, 0, stream>>>(c2w, W3, 192, 192, 128 * 9 * 192);
  wxform_kernel<<<blocks(128LL * 9 * 256), 256, 0, stream>>>(c3w1, W4, 240, 256, 128 * 9 * 256);
  wxform_kernel<<<blocks(256LL * 9 * 128), 256, 0, stream>>>(c3w2, W5, 128, 128, 256 * 9 * 128);

  // 2) stem conv + style1 concat
  conv0_kernel<<<blocks(2LL * 58 * 64 * 512), 256, 0, stream>>>(x, in_w, in_b, A0,
                                                                2 * 58 * 64 * 512);
  fill_style1<<<blocks(2LL * 58 * 512 * 32), 256, 0, stream>>>(SB, A0, 2 * 58 * 512 * 32);

  // 3) c1w1: A0(96ch) -> B1 (2,56,512,64)
  conv3x3_wmma<<<dim3(32, 56, 2), dim3(128), 0, stream>>>(A0, W1, c1b1, B1,
      58, 514, 96, 56, 512, 64, 512, 64, 0);
  // pool -> A1 (2,28,258,64)
  pool2_kernel<<<blocks(2LL * 28 * 256 * 64), 256, 0, stream>>>(B1, A1,
      28, 256, 64, 512, 258, 64, 1, 2 * 28 * 256 * 64);

  // 4) c1w2: A1 -> A2 ch 0..127 ; style2 -> ch 128..191
  conv3x3_wmma<<<dim3(16, 26, 2), dim3(256), 0, stream>>>(A1, W2, c1b2, A2,
      28, 258, 64, 26, 256, 128, 258, 192, 1);
  fill_style2<<<blocks(2LL * 26 * 256 * 64), 256, 0, stream>>>(SB, A2, 2 * 26 * 256 * 64);

  // 5) c2w: A2(192ch) -> B2 (2,24,256,128) ; pool -> A3 ch 0..127 ; sal -> 128..239
  conv3x3_wmma<<<dim3(16, 24, 2), dim3(256), 0, stream>>>(A2, W3, c2b, B2,
      26, 258, 192, 24, 256, 128, 256, 128, 0);
  pool2_kernel<<<blocks(2LL * 12 * 128 * 128), 256, 0, stream>>>(B2, A3,
      12, 128, 128, 256, 130, 256, 1, 2 * 12 * 128 * 128);
  fill_sal<<<blocks(2LL * 12 * 128 * 112), 256, 0, stream>>>(sstyle, label, A3,
      2 * 12 * 128 * 112);

  // 6) c3w1: A3(256ch padded) -> B3 (2,10,128,128) ; pool -> A4
  conv3x3_wmma<<<dim3(8, 10, 2), dim3(256), 0, stream>>>(A3, W4, c3b1, B3,
      12, 130, 256, 10, 128, 128, 128, 128, 0);
  pool2_kernel<<<blocks(2LL * 5 * 64 * 128), 256, 0, stream>>>(B3, A4,
      5, 64, 128, 128, 66, 128, 1, 2 * 5 * 64 * 128);

  // 7) c3w2: A4 -> A5 (2,3,68,256) with wpad 2 (== mLp)
  conv3x3_wmma<<<dim3(4, 3, 2), dim3(512), 0, stream>>>(A4, W5, c3b2, A5,
      5, 66, 128, 3, 64, 256, 68, 256, 2);

  // 8) heads
  fm_kernel<<<dim3(128), dim3(256), 0, stream>>>(A5, fm_w, fm_b, out);       // pM
  fill_cs<<<blocks(256LL * 32), 256, 0, stream>>>(cstyle, CL, FT, 256 * 32);
  cd_conv_wmma<<<dim3(256), dim3(256), 0, stream>>>(A5, cdw, cdb, CL, FT);
  fc_kernel<<<dim3(256), dim3(128), 0, stream>>>(FT, CL, f1w, f1b, f2w, f2b, out); // pChar
}
